// GraphConvNet_66795331387690
// MI455X (gfx1250) — compile-verified, matrix operand order checked by reading, not run on hardware
//
#include <hip/hip_runtime.h>
#include <hip/hip_bf16.h>
#include <stdint.h>

// ---------------------------------------------------------------------------
// GCN (3x GCNConv + BN + ReLU) -> mean/max pool -> 2-layer MLP head
// MI455X / gfx1250: GEMMs via v_wmma_f32_16x16x32_f16 (wave32, 16x16 tiles)
// ---------------------------------------------------------------------------

typedef __attribute__((ext_vector_type(16))) _Float16 v16h;
typedef __attribute__((ext_vector_type(8)))  float    v8f;

#define NNODES 100000
#define NEDGES 1600000
#define NGRAPH 128
#define INDIM  32
#define HIDDIM 128
#define BN_EPS 1e-5f

union frag16 { uint4 u[2]; v16h h; };

// ---------------------------- utility kernels ------------------------------

__global__ void k_fill_f32(float* __restrict__ p, int n, float v) {
  int i = blockIdx.x * blockDim.x + threadIdx.x;
  if (i < n) p[i] = v;
}

__global__ void k_f32_to_f16(const float* __restrict__ in, _Float16* __restrict__ out, int n) {
  int i = blockIdx.x * blockDim.x + threadIdx.x;
  if (i < n) out[i] = (_Float16)in[i];
}

// W is [K,M] row-major (f32); write transposed f16 [M,K] so the WMMA B
// fragment loads become contiguous 16B chunks per lane (same as A).
__global__ void k_weight_t_f16(const float* __restrict__ W, _Float16* __restrict__ wT,
                               int K, int M) {
  int i = blockIdx.x * blockDim.x + threadIdx.x;
  if (i < K * M) {
    int k = i / M, m = i % M;
    wT[(size_t)m * K + k] = (_Float16)W[i];
  }
}

// ------------------------------ degree / norm ------------------------------

__global__ void k_deg_accum(const int* __restrict__ dst, float* __restrict__ deg, int E) {
  int e = blockIdx.x * blockDim.x + threadIdx.x;
  if (e < E) atomicAdd(&deg[dst[e]], 1.0f);
}

__global__ void k_rsqrt_inplace(float* __restrict__ p, int n) {
  int i = blockIdx.x * blockDim.x + threadIdx.x;
  if (i < n) p[i] = rsqrtf(p[i]);
}

// ------------------------------- WMMA GEMM ---------------------------------
// C[nrows,M] = A[nrows,K] (f16, row-major) * Bt[M,K]^T (f16)
// One wave computes one 16x16 output tile; 8 waves per 256-thread block.
// nrows % 16 == 0, M % 16 == 0, K % 32 == 0 (all hold for this net).
__global__ __launch_bounds__(256)
void k_gemm_wmma_f16(const _Float16* __restrict__ A, const _Float16* __restrict__ Bt,
                     float* __restrict__ C, int nrows, int K, int M) {
  int wave = blockIdx.x * 8 + (threadIdx.x >> 5);
  int lane = threadIdx.x & 31;
  int tilesM = M >> 4;
  int ntiles = (nrows >> 4) * tilesM;
  if (wave >= ntiles) return;                 // wave-uniform: EXEC stays all-ones
  int tr = wave / tilesM;
  int tc = wave % tilesM;

  int half16 = (lane >> 4) << 3;              // K sub-chunk select per A/B layout
  const _Float16* arow = A  + (size_t)(tr * 16 + (lane & 15)) * K + half16;
  const _Float16* brow = Bt + (size_t)(tc * 16 + (lane & 15)) * K + half16;

  v8f acc = {};
  for (int kk = 0; kk < K; kk += 32) {
    frag16 a, b;
    a.u[0] = *(const uint4*)(arow + kk);       // K = kk + half16 + [0..7]
    a.u[1] = *(const uint4*)(arow + kk + 16);  // K = kk + 16 + half16 + [0..7]
    b.u[0] = *(const uint4*)(brow + kk);
    b.u[1] = *(const uint4*)(brow + kk + 16);
    acc = __builtin_amdgcn_wmma_f32_16x16x32_f16(
        false, a.h, false, b.h, (short)0, acc, false, false);
  }

  // D layout: VGPR i -> M = i + 8*(lane/16), N = lane%16
  int mrow = tr * 16 + ((lane >> 4) << 3);
  int col  = tc * 16 + (lane & 15);
#pragma unroll
  for (int i = 0; i < 8; ++i)
    C[(size_t)(mrow + i) * M + col] = acc[i];
}

// --------------------------- GCN aggregation -------------------------------

// agg[v,c] = xw[v,c] * dinv[v]^2 + b[c]   (self-loop term + bias)
__global__ void k_selfloop_bias(const float* __restrict__ xw, const float* __restrict__ dinv,
                                const float* __restrict__ b, float* __restrict__ agg,
                                int n, int M) {
  int i = blockIdx.x * blockDim.x + threadIdx.x;
  if (i < n * M) {
    int r = i / M, c = i % M;
    float dv = dinv[r];
    agg[i] = xw[i] * dv * dv + b[c];
  }
}

// For each edge: agg[dst] += xw[src] * dinv[src] * dinv[dst]
// (M/4) threads per edge, float4 gather + 4x f32 atomic scatter.
__global__ void k_edge_scatter(const int* __restrict__ src, const int* __restrict__ dst,
                               const float* __restrict__ dinv, const float* __restrict__ xw,
                               float* __restrict__ agg, int E, int M) {
  int lanes = M >> 2;
  int idx = blockIdx.x * blockDim.x + threadIdx.x;
  int e = idx / lanes;
  if (e >= E) return;
  int c = (idx % lanes) << 2;
  int s = src[e], d = dst[e];
  float nrm = dinv[s] * dinv[d];
  float4 v = *(const float4*)(xw + (size_t)s * M + c);
  float* a = agg + (size_t)d * M + c;
  atomicAdd(a + 0, v.x * nrm);
  atomicAdd(a + 1, v.y * nrm);
  atomicAdd(a + 2, v.z * nrm);
  atomicAdd(a + 3, v.w * nrm);
}

// ------------------------------- batchnorm ---------------------------------

// blockDim.x == M; each thread owns one column, strided over rows.
__global__ void k_bn_stats(const float* __restrict__ h, float* __restrict__ stats,
                           int n, int M) {
  int c = threadIdx.x;
  float s = 0.f, s2 = 0.f;
  for (int r = blockIdx.x; r < n; r += gridDim.x) {
    float v = h[(size_t)r * M + c];
    s += v; s2 += v * v;
  }
  atomicAdd(&stats[c], s);
  atomicAdd(&stats[M + c], s2);
}

__global__ void k_bn_final(float* __restrict__ stats, const float* __restrict__ g,
                           const float* __restrict__ be, int n, int M) {
  int c = blockIdx.x * blockDim.x + threadIdx.x;
  if (c < M) {
    float mu  = stats[c] / (float)n;
    float var = stats[M + c] / (float)n - mu * mu;
    float sc  = g[c] * rsqrtf(var + BN_EPS);
    stats[c]     = sc;
    stats[M + c] = be[c] - mu * sc;
  }
}

// y = relu(x*scale + shift); write f32 (pooling) and f16 (next GEMM input)
__global__ void k_bn_apply(const float* __restrict__ h, const float* __restrict__ stats,
                           float* __restrict__ out32, _Float16* __restrict__ out16,
                           int n, int M) {
  int i = blockIdx.x * blockDim.x + threadIdx.x;
  if (i < n * M) {
    int c = i % M;
    float v = h[i] * stats[c] + stats[M + c];
    v = v > 0.f ? v : 0.f;
    out32[i] = v;
    out16[i] = (_Float16)v;
  }
}

// -------------------------------- pooling ----------------------------------

// h: [n,64] (post-ReLU, >= 0). 16 threads of 4 channels per node.
// max via int-bit atomicMax (valid for non-negative floats, init 0).
__global__ void k_pool(const float* __restrict__ h, const int* __restrict__ batch,
                       float* __restrict__ sum, float* __restrict__ mx,
                       float* __restrict__ cnt, int n) {
  int idx = blockIdx.x * blockDim.x + threadIdx.x;
  int node = idx >> 4;
  if (node >= n) return;
  int c = (idx & 15) << 2;
  int g = batch[node];
  float4 v = *(const float4*)(h + (size_t)node * 64 + c);
  float* sp = sum + g * 64 + c;
  atomicAdd(sp + 0, v.x); atomicAdd(sp + 1, v.y);
  atomicAdd(sp + 2, v.z); atomicAdd(sp + 3, v.w);
  int* mp = (int*)(mx + g * 64 + c);
  atomicMax(mp + 0, __float_as_int(v.x));
  atomicMax(mp + 1, __float_as_int(v.y));
  atomicMax(mp + 2, __float_as_int(v.z));
  atomicMax(mp + 3, __float_as_int(v.w));
  if (c == 0) atomicAdd(&cnt[g], 1.0f);
}

__global__ void k_gfeat(const float* __restrict__ sum, const float* __restrict__ mx,
                        const float* __restrict__ cnt, float* __restrict__ gf) {
  int i = blockIdx.x * blockDim.x + threadIdx.x;
  if (i < NGRAPH * HIDDIM) {
    int g = i / HIDDIM, j = i % HIDDIM;
    gf[i] = (j < 64) ? sum[g * 64 + j] / fmaxf(cnt[g], 1.0f) : mx[g * 64 + (j - 64)];
  }
}

// ------------------------------- MLP head ----------------------------------

__global__ void k_head1(const float* __restrict__ gf, const float* __restrict__ Wc1,
                        const float* __restrict__ bc1, float* __restrict__ hid) {
  int i = blockIdx.x * blockDim.x + threadIdx.x;          // 128*64
  if (i < NGRAPH * 64) {
    int g = i / 64, j = i % 64;
    float acc = bc1[j];
    for (int k = 0; k < HIDDIM; ++k) acc += gf[g * HIDDIM + k] * Wc1[k * 64 + j];
    hid[i] = fmaxf(acc, 0.f);
  }
}

__global__ void k_head2(const float* __restrict__ hid, const float* __restrict__ Wc2,
                        const float* __restrict__ bc2, float* __restrict__ out) {
  int i = blockIdx.x * blockDim.x + threadIdx.x;          // 128*2
  if (i < NGRAPH * 2) {
    int g = i / 2, o = i % 2;
    float acc = bc2[o];
    for (int j = 0; j < 64; ++j) acc += hid[g * 64 + j] * Wc2[j * 2 + o];
    out[i] = acc;
  }
}

// -------------------------------- driver -----------------------------------

static inline int cdiv(long long a, int b) { return (int)((a + b - 1) / b); }

extern "C" void kernel_launch(void* const* d_in, const int* in_sizes, int n_in,
                              void* d_out, int out_size, void* d_ws, size_t ws_size,
                              hipStream_t stream) {
  (void)in_sizes; (void)n_in; (void)out_size; (void)ws_size;

  const float* x    = (const float*)d_in[0];
  const int*   ei   = (const int*)d_in[1];
  const int*   batch= (const int*)d_in[2];
  const float* W1   = (const float*)d_in[3];  const float* b1  = (const float*)d_in[4];
  const float* g1   = (const float*)d_in[5];  const float* be1 = (const float*)d_in[6];
  const float* W2   = (const float*)d_in[7];  const float* b2  = (const float*)d_in[8];
  const float* g2   = (const float*)d_in[9];  const float* be2 = (const float*)d_in[10];
  const float* W3   = (const float*)d_in[11]; const float* b3  = (const float*)d_in[12];
  const float* g3   = (const float*)d_in[13]; const float* be3 = (const float*)d_in[14];
  const float* Wc1  = (const float*)d_in[15]; const float* bc1 = (const float*)d_in[16];
  const float* Wc2  = (const float*)d_in[17]; const float* bc2 = (const float*)d_in[18];
  float* out = (float*)d_out;

  const int* src = ei;
  const int* dst = ei + NEDGES;

  // workspace carve-up (256B aligned)
  char* ws = (char*)d_ws;
  size_t off = 0;
  auto carve = [&](size_t bytes) {
    size_t o = off;
    off = (off + bytes + 255) & ~(size_t)255;
    return (void*)(ws + o);
  };
  float*    dinv  = (float*)   carve((size_t)NNODES * 4);
  float*    bufA  = (float*)   carve((size_t)NNODES * HIDDIM * 4);  // GEMM out / BN'd f32
  float*    bufB  = (float*)   carve((size_t)NNODES * HIDDIM * 4);  // aggregation
  _Float16* bufH  = (_Float16*)carve((size_t)NNODES * HIDDIM * 2);  // f16 activations
  _Float16* wT    = (_Float16*)carve((size_t)HIDDIM * HIDDIM * 2);  // f16 W^T
  float*    stats = (float*)   carve(2 * HIDDIM * 4);
  float*    psum  = (float*)   carve((size_t)NGRAPH * 64 * 4);
  float*    pmax  = (float*)   carve((size_t)NGRAPH * 64 * 4);
  float*    pcnt  = (float*)   carve((size_t)NGRAPH * 4);
  float*    gf    = (float*)   carve((size_t)NGRAPH * HIDDIM * 4);
  float*    hid   = (float*)   carve((size_t)NGRAPH * 64 * 4);

  const int T = 256;

  // --- degree & symmetric norm (deg = in-degree + 1 from self loop) ---
  k_fill_f32<<<cdiv(NNODES, T), T, 0, stream>>>(dinv, NNODES, 1.0f);
  k_deg_accum<<<cdiv(NEDGES, T), T, 0, stream>>>(dst, dinv, NEDGES);
  k_rsqrt_inplace<<<cdiv(NNODES, T), T, 0, stream>>>(dinv, NNODES);

  // --- layer 0 input to f16 ---
  k_f32_to_f16<<<cdiv((long long)NNODES * INDIM, T), T, 0, stream>>>(x, bufH, NNODES * INDIM);

  struct Layer { const float *W, *b, *g, *be; int K, M; };
  Layer layers[3] = {
    { W1, b1, g1, be1, INDIM,  HIDDIM },
    { W2, b2, g2, be2, HIDDIM, HIDDIM },
    { W3, b3, g3, be3, HIDDIM, HIDDIM / 2 },
  };

  for (int L = 0; L < 3; ++L) {
    int K = layers[L].K, M = layers[L].M;
    // weight -> f16 transpose
    k_weight_t_f16<<<cdiv(K * M, T), T, 0, stream>>>(layers[L].W, wT, K, M);
    // xw = act @ W  (WMMA)
    int ntiles = (NNODES / 16) * (M / 16);
    k_gemm_wmma_f16<<<cdiv(ntiles, 8), T, 0, stream>>>(bufH, wT, bufA, NNODES, K, M);
    // self-loop + bias, then edge scatter
    k_selfloop_bias<<<cdiv((long long)NNODES * M, T), T, 0, stream>>>(bufA, dinv, layers[L].b, bufB, NNODES, M);
    k_edge_scatter<<<cdiv((long long)NEDGES * (M / 4), T), T, 0, stream>>>(src, dst, dinv, bufA, bufB, NEDGES, M);
    // batchnorm + relu
    k_fill_f32<<<cdiv(2 * M, T), T, 0, stream>>>(stats, 2 * M, 0.0f);
    k_bn_stats<<<2048, M, 0, stream>>>(bufB, stats, NNODES, M);
    k_bn_final<<<1, M, 0, stream>>>(stats, layers[L].g, layers[L].be, NNODES, M);
    k_bn_apply<<<cdiv((long long)NNODES * M, T), T, 0, stream>>>(bufB, stats, bufA, bufH, NNODES, M);
  }

  // --- pooling over graphs (h = bufA, [N,64]) ---
  k_fill_f32<<<cdiv(NGRAPH * 64, T), T, 0, stream>>>(psum, NGRAPH * 64, 0.0f);
  k_fill_f32<<<cdiv(NGRAPH * 64, T), T, 0, stream>>>(pmax, NGRAPH * 64, 0.0f);
  k_fill_f32<<<1, NGRAPH, 0, stream>>>(pcnt, NGRAPH, 0.0f);
  k_pool<<<cdiv((long long)NNODES * 16, T), T, 0, stream>>>(bufA, batch, psum, pmax, pcnt, NNODES);
  k_gfeat<<<cdiv(NGRAPH * HIDDIM, T), T, 0, stream>>>(psum, pmax, pcnt, gf);

  // --- MLP head ---
  k_head1<<<cdiv(NGRAPH * 64, T), T, 0, stream>>>(gf, Wc1, bc1, hid);
  k_head2<<<1, NGRAPH * 2, 0, stream>>>(hid, Wc2, bc2, out);
}
